// Decoder_16965120819493
// MI455X (gfx1250) — compile-verified, matrix operand order checked by reading, not run on hardware
//
#include <hip/hip_runtime.h>

// ---------------------------------------------------------------------------
// MI455X (gfx1250) persistent Tacotron-decoder.
// wave32, WMMA 16x16x32 bf16 for all big GEMMs, weights staged to bf16 in L2.
// ---------------------------------------------------------------------------

#define NWG    64
#define NTHR   256
#define NWAVES (NWG * NTHR / 32)   // 512 waves
#define GTHREADS (NWG * NTHR)      // 16384 threads

typedef __attribute__((ext_vector_type(16))) __bf16 v16bf;
typedef __attribute__((ext_vector_type(8)))  float  v8f;

union Frag { v16bf v; unsigned u[8]; };

struct Params {
  // inputs
  const float *memory, *decoder_inputs, *z_l, *z_o;
  const int   *mem_len;
  const float *pre_w1, *pre_w2;
  const float *a_wih, *a_whh, *a_bih, *a_bhh;
  const float *wq, *wm, *vvec, *loc_conv, *loc_dense;
  const float *d_wih, *d_whh, *d_bih, *d_bhh;
  const float *res_w, *res_b, *proj_w, *proj_b, *gate_w, *gate_b;
  // workspace: control + fp32 state (zeroed each call)
  unsigned *ctrl;
  float *att_h, *att_c, *dec_h, *dec_c, *aw, *aw_cum, *ctx, *dec_in_f;
  // workspace: fp32 scratch
  float *gates, *energ, *pq, *pa, *pm;
  // workspace: bf16 (ushort) state (zeroed) + scratch + weights
  unsigned short *xcat, *att_h_bf, *dec_h_bf, *dec_in_bf;
  unsigned short *loc_bf, *resid_bf, *pre_bf;
  unsigned short *wq_bf, *ldense_bf, *a_wih_bf, *a_whh_bf;
  unsigned short *d_wih_bf, *d_whh_bf, *res_w_bf, *proj_w_bf;
  // outputs
  float *out_mel, *out_gate, *out_align;
};

// ---------------- helpers ----------------

__device__ __forceinline__ unsigned short f2bf(float f) {
  unsigned u = __float_as_uint(f);
  unsigned r = u + 0x7FFFu + ((u >> 16) & 1u);   // round-to-nearest-even
  return (unsigned short)(r >> 16);
}

__device__ __forceinline__ float sigm(float x) {
  return 1.0f / (1.0f + __expf(-x));
}

__device__ __forceinline__ float rng01(unsigned a, unsigned b) {
  unsigned x = a * 0x9E3779B9u + b * 0x85EBCA6Bu + 0x27D4EB2Fu;
  x ^= x >> 16; x *= 0x7FEB352Du; x ^= x >> 15; x *= 0x846CA68Bu; x ^= x >> 16;
  return (float)(x >> 8) * (1.0f / 16777216.0f);
}

// Load one 16x32 bf16 WMMA operand fragment.
// Row-major source: element (row, k).  Per ISA 7.12.2 16-bit A layout:
// lane (l%16)=row, half=l/16;  VGPR v in 0..3 -> k = half*8 + 2v,
// VGPR v in 4..7 -> k = 16 + half*8 + 2(v-4).  B uses the mirrored layout
// (col across lanes), so the same loader works with row := weight row (= N).
__device__ __forceinline__ v16bf load_frag(const unsigned short* base,
                                           int row, int ld, int k0, int half) {
  Frag f;
  const unsigned short* p = base + (size_t)row * ld + k0 + half * 8;
#pragma unroll
  for (int i = 0; i < 4; ++i) f.u[i]     = *(const unsigned*)(p + 2 * i);
#pragma unroll
  for (int i = 0; i < 4; ++i) f.u[4 + i] = *(const unsigned*)(p + 16 + 2 * i);
  return f.v;
}

__device__ __forceinline__ v8f wmma_bf16(v16bf a, v16bf b, v8f c) {
  return __builtin_amdgcn_wmma_f32_16x16x32_bf16(false, a, false, b,
                                                 (short)0, c, false, false);
}

__device__ __forceinline__ void grid_barrier(unsigned* cnt, unsigned* gen,
                                             unsigned nwg) {
  __syncthreads();
  if (threadIdx.x == 0) {
    __threadfence();
    volatile unsigned* vgen = (volatile unsigned*)gen;
    unsigned g = *vgen;
    if (atomicAdd(cnt, 1u) == nwg - 1u) {
      *((volatile unsigned*)cnt) = 0u;
      __threadfence();
      atomicAdd(gen, 1u);
    } else {
      while (*vgen == g) __builtin_amdgcn_s_sleep(2);
    }
    __threadfence();
  }
  __syncthreads();
}

// ---------------- prologue kernels ----------------

__global__ void convert_kernel(const float* s, unsigned short* d, int n) {
  for (int i = blockIdx.x * blockDim.x + threadIdx.x; i < n;
       i += gridDim.x * blockDim.x)
    d[i] = f2bf(s[i]);
}

// Prenet over all 500 frames: two dropout(0.5) ReLU layers -> bf16.
__global__ void prenet_kernel(Params P) {
  __shared__ float h1[256];
  int t = blockIdx.x >> 5;       // 0..499
  int b = blockIdx.x & 31;
  int j = threadIdx.x;
  float acc = 0.f;
  if (t > 0) {
    const float* di = P.decoder_inputs + (size_t)b * 80 * 500 + (t - 1);
    const float* w  = P.pre_w1 + j * 80;
    for (int k = 0; k < 80; ++k) acc += di[k * 500] * w[k];
  }
  acc = fmaxf(acc, 0.f);
  acc = (rng01(0x1111u + t, b * 256 + j) < 0.5f) ? acc * 2.0f : 0.f;
  h1[j] = acc;
  __syncthreads();
  float a2 = 0.f;
  const float* w2 = P.pre_w2 + j * 256;
  for (int k = 0; k < 256; ++k) a2 += h1[k] * w2[k];
  a2 = fmaxf(a2, 0.f);
  a2 = (rng01(0x2222u + t, b * 256 + j) < 0.5f) ? a2 * 2.0f : 0.f;
  P.pre_bf[((size_t)t * 32 + b) * 256 + j] = f2bf(a2);
}

// processed_memory[b,ti,a] = memory[b,ti,:] . wm[a,:]
__global__ void pm_kernel(Params P) {
  int NT = gridDim.x * blockDim.x;
  for (int i = blockIdx.x * blockDim.x + threadIdx.x; i < 32 * 256 * 128;
       i += NT) {
    int bt = i >> 7, a = i & 127;
    const float* m = P.memory + (size_t)bt * 512;
    const float* w = P.wm + a * 512;
    float s = 0.f;
    for (int k = 0; k < 512; ++k) s += m[k] * w[k];
    P.pm[i] = s;
  }
}

// Constant pieces: z -> decoder-input tail, pre[0] -> attention xcat.
__global__ void misc_kernel(Params P) {
  int i = blockIdx.x * blockDim.x + threadIdx.x;
  if (i < 32 * 64) {
    int b = i >> 6, j = i & 63;
    float zv = (j < 32) ? P.z_l[b * 32 + j] : P.z_o[b * 32 + (j - 32)];
    P.dec_in_f[b * 1600 + 1536 + j]  = zv;
    P.dec_in_bf[b * 1600 + 1536 + j] = f2bf(zv);
  }
  if (i < 32 * 256) {
    int b = i >> 8, k = i & 255;
    P.xcat[b * 768 + k] = P.pre_bf[(size_t)b * 256 + k];
  }
}

// ---------------- persistent decoder ----------------

__global__ void __launch_bounds__(NTHR, 1) decoder_kernel(Params P) {
  const int tid    = blockIdx.x * NTHR + threadIdx.x;
  const int lane   = threadIdx.x & 31;
  const int gwave  = tid >> 5;
  const int lane15 = lane & 15;
  const int half   = lane >> 4;
  unsigned* cnt = P.ctrl;
  unsigned* gen = P.ctrl + 1;

  for (int t = 0; t < 500; ++t) {
    // ---- P1: attention LSTM gates [32,4096] via WMMA -------------------
    {
      int mt = gwave >> 8, nt = gwave & 255;           // 2 x 256 tiles
      int arow = mt * 16 + lane15;
      int brow = nt * 16 + lane15;
      v8f acc = {};
      for (int k0 = 0; k0 < 768; k0 += 32)
        acc = wmma_bf16(load_frag(P.xcat, arow, 768, k0, half),
                        load_frag(P.a_wih_bf, brow, 768, k0, half), acc);
      for (int k0 = 0; k0 < 1024; k0 += 32)
        acc = wmma_bf16(load_frag(P.att_h_bf, arow, 1024, k0, half),
                        load_frag(P.a_whh_bf, brow, 1024, k0, half), acc);
      int n = nt * 16 + lane15;
#pragma unroll
      for (int r = 0; r < 8; ++r)
        P.gates[(mt * 16 + r + 8 * half) * 4096 + n] = acc[r];
    }
    grid_barrier(cnt, gen, NWG);

    // ---- P2: attention LSTM pointwise + location conv ------------------
    for (int i = tid; i < 32 * 1024; i += GTHREADS) {
      int b = i >> 10, j = i & 1023;
      const float* g = P.gates + b * 4096;
      float gi = g[j]        + P.a_bih[j]        + P.a_bhh[j];
      float gf = g[1024 + j] + P.a_bih[1024 + j] + P.a_bhh[1024 + j];
      float gg = g[2048 + j] + P.a_bih[2048 + j] + P.a_bhh[2048 + j];
      float go = g[3072 + j] + P.a_bih[3072 + j] + P.a_bhh[3072 + j];
      float c = sigm(gf) * P.att_c[i] + sigm(gi) * tanhf(gg);
      float h = sigm(go) * tanhf(c);
      h = (rng01(0x1234u + (unsigned)t, (unsigned)i) < 0.9f)
              ? h * (1.0f / 0.9f) : 0.0f;
      P.att_c[i] = c;
      P.att_h[i] = h;
      unsigned short hb = f2bf(h);
      P.att_h_bf[i] = hb;
      P.dec_in_bf[b * 1600 + j] = hb;
      P.dec_in_f[b * 1600 + j]  = h;
    }
    for (int i = tid; i < 32 * 32 * 256; i += GTHREADS) {
      int b = i >> 13, f = (i >> 8) & 31, ti = i & 255;
      float s = 0.f;
      const float* w = P.loc_conv + f * 62;
      for (int k = 0; k < 31; ++k) {
        int p = ti - 15 + k;
        if (p >= 0 && p < 256) {
          s += P.aw[b * 256 + p]     * w[k];
          s += P.aw_cum[b * 256 + p] * w[31 + k];
        }
      }
      P.loc_bf[(b * 256 + ti) * 32 + f] = f2bf(s);
    }
    grid_barrier(cnt, gen, NWG);

    // ---- P3: proc_att [8192,128] (K=32) and pq [32,128] via WMMA -------
    for (int job = gwave; job < 4096 + 16; job += NWAVES) {
      if (job < 4096) {
        int mt = job >> 3, nt = job & 7;
        v8f acc = {};
        acc = wmma_bf16(load_frag(P.loc_bf, mt * 16 + lane15, 32, 0, half),
                        load_frag(P.ldense_bf, nt * 16 + lane15, 32, 0, half),
                        acc);
        int n = nt * 16 + lane15;
#pragma unroll
        for (int r = 0; r < 8; ++r)
          P.pa[(size_t)(mt * 16 + r + 8 * half) * 128 + n] = acc[r];
      } else {
        int j2 = job - 4096;
        int mt = j2 >> 3, nt = j2 & 7;
        v8f acc = {};
        for (int k0 = 0; k0 < 1024; k0 += 32)
          acc = wmma_bf16(load_frag(P.att_h_bf, mt * 16 + lane15, 1024, k0, half),
                          load_frag(P.wq_bf, nt * 16 + lane15, 1024, k0, half),
                          acc);
        int n = nt * 16 + lane15;
#pragma unroll
        for (int r = 0; r < 8; ++r)
          P.pq[(mt * 16 + r + 8 * half) * 128 + n] = acc[r];
      }
    }
    grid_barrier(cnt, gen, NWG);

    // ---- P4: energies = v . tanh(pq + pa + pm), mask pads --------------
    for (int i = tid; i < 32 * 256; i += GTHREADS) {
      int b = i >> 8, ti = i & 255;
      float e;
      if (ti >= P.mem_len[b]) {
        e = -1e9f;
      } else {
        e = 0.f;
        const float* q  = P.pq + b * 128;
        const float* pa = P.pa + (size_t)i * 128;
        const float* pm = P.pm + (size_t)i * 128;
        for (int a = 0; a < 128; ++a)
          e += P.vvec[a] * tanhf(q[a] + pa[a] + pm[a]);
      }
      P.energ[i] = e;
    }
    grid_barrier(cnt, gen, NWG);

    // ---- P5: softmax, aw_cum, alignments out, context, next xcat -------
    if (gwave < 32) {
      int b = gwave;
      float er[8];
#pragma unroll
      for (int j = 0; j < 8; ++j) er[j] = P.energ[b * 256 + lane + 32 * j];
      float mx = er[0];
#pragma unroll
      for (int j = 1; j < 8; ++j) mx = fmaxf(mx, er[j]);
      for (int m = 16; m >= 1; m >>= 1) mx = fmaxf(mx, __shfl_xor(mx, m, 32));
      float s = 0.f;
#pragma unroll
      for (int j = 0; j < 8; ++j) { er[j] = __expf(er[j] - mx); s += er[j]; }
      for (int m = 16; m >= 1; m >>= 1) s += __shfl_xor(s, m, 32);
      float inv = 1.0f / s;
#pragma unroll
      for (int j = 0; j < 8; ++j) {
        float a = er[j] * inv;
        er[j] = a;
        int ti = lane + 32 * j;
        P.aw[b * 256 + ti] = a;
        float cum = P.aw_cum[b * 256 + ti] + a;
        P.aw_cum[b * 256 + ti] = cum;
        P.out_align[(size_t)(b * 500 + t) * 256 + ti] = a;
      }
      float acc[16];
#pragma unroll
      for (int q = 0; q < 16; ++q) acc[q] = 0.f;
      for (int ti = 0; ti < 256; ++ti) {
        float a = __shfl(er[ti >> 5], ti & 31, 32);
        const float* m = P.memory + (size_t)(b * 256 + ti) * 512 + lane;
#pragma unroll
        for (int q = 0; q < 16; ++q) acc[q] += a * m[q * 32];
      }
#pragma unroll
      for (int q = 0; q < 16; ++q) {
        int e = lane + q * 32;
        P.ctx[b * 512 + e] = acc[q];
        unsigned short hb = f2bf(acc[q]);
        P.xcat[b * 768 + 256 + e]      = hb;
        P.dec_in_bf[b * 1600 + 1024 + e] = hb;
        P.dec_in_f[b * 1600 + 1024 + e]  = acc[q];
      }
      int tn = (t + 1 < 500) ? t + 1 : 499;   // prenet frame for next step
      for (int k = lane; k < 256; k += 32)
        P.xcat[b * 768 + k] = P.pre_bf[((size_t)tn * 32 + b) * 256 + k];
    }
    grid_barrier(cnt, gen, NWG);

    // ---- P6: decoder LSTM gates [32,4096] via WMMA ---------------------
    {
      int mt = gwave >> 8, nt = gwave & 255;
      int arow = mt * 16 + lane15;
      int brow = nt * 16 + lane15;
      v8f acc = {};
      for (int k0 = 0; k0 < 1600; k0 += 32)
        acc = wmma_bf16(load_frag(P.dec_in_bf, arow, 1600, k0, half),
                        load_frag(P.d_wih_bf, brow, 1600, k0, half), acc);
      for (int k0 = 0; k0 < 1024; k0 += 32)
        acc = wmma_bf16(load_frag(P.dec_h_bf, arow, 1024, k0, half),
                        load_frag(P.d_whh_bf, brow, 1024, k0, half), acc);
      int n = nt * 16 + lane15;
#pragma unroll
      for (int r = 0; r < 8; ++r)
        P.gates[(mt * 16 + r + 8 * half) * 4096 + n] = acc[r];
    }
    grid_barrier(cnt, gen, NWG);

    // ---- P7: decoder LSTM pointwise ------------------------------------
    for (int i = tid; i < 32 * 1024; i += GTHREADS) {
      int b = i >> 10, j = i & 1023;
      const float* g = P.gates + b * 4096;
      float gi = g[j]        + P.d_bih[j]        + P.d_bhh[j];
      float gf = g[1024 + j] + P.d_bih[1024 + j] + P.d_bhh[1024 + j];
      float gg = g[2048 + j] + P.d_bih[2048 + j] + P.d_bhh[2048 + j];
      float go = g[3072 + j] + P.d_bih[3072 + j] + P.d_bhh[3072 + j];
      float c = sigm(gf) * P.dec_c[i] + sigm(gi) * tanhf(gg);
      float h = sigm(go) * tanhf(c);
      h = (rng01(0x9876u + (unsigned)t, 0x100000u + (unsigned)i) < 0.9f)
              ? h * (1.0f / 0.9f) : 0.0f;
      P.dec_c[i] = c;
      P.dec_h[i] = h;
      P.dec_h_bf[i] = f2bf(h);
    }
    grid_barrier(cnt, gen, NWG);

    // ---- P8: resid = dec_in + dec_h @ res_w^T + res_b (WMMA) -----------
    if (gwave < 200) {
      int mt = gwave / 100, nt = gwave % 100;
      int n = nt * 16 + lane15;
      v8f acc;
#pragma unroll
      for (int r = 0; r < 8; ++r)
        acc[r] = P.dec_in_f[(mt * 16 + r + 8 * half) * 1600 + n] + P.res_b[n];
      for (int k0 = 0; k0 < 1024; k0 += 32)
        acc = wmma_bf16(load_frag(P.dec_h_bf, mt * 16 + lane15, 1024, k0, half),
                        load_frag(P.res_w_bf, n, 1024, k0, half), acc);
#pragma unroll
      for (int r = 0; r < 8; ++r)
        P.resid_bf[(mt * 16 + r + 8 * half) * 1600 + n] = f2bf(acc[r]);
    }
    grid_barrier(cnt, gen, NWG);

    // ---- P9: mel = resid @ proj_w^T + proj_b (WMMA); stop-gate ---------
    if (gwave < 10) {
      int mt = gwave / 5, nt = gwave % 5;
      v8f acc = {};
      for (int k0 = 0; k0 < 1600; k0 += 32)
        acc = wmma_bf16(load_frag(P.resid_bf, mt * 16 + lane15, 1600, k0, half),
                        load_frag(P.proj_w_bf, nt * 16 + lane15, 1600, k0, half),
                        acc);
      int n = nt * 16 + lane15;
#pragma unroll
      for (int r = 0; r < 8; ++r) {
        int b = mt * 16 + r + 8 * half;
        P.out_mel[(size_t)(b * 80 + n) * 500 + t] = acc[r] + P.proj_b[n];
      }
    } else if (gwave == 16) {
      int b = lane;
      float s = P.gate_b[0];
      for (int j = 0; j < 1024; ++j) s += P.dec_h[b * 1024 + j] * P.gate_w[j];
      for (int e = 0; e < 512; ++e)  s += P.ctx[b * 512 + e] * P.gate_w[1024 + e];
      P.out_gate[b * 500 + t] = s;
    }
    // No barrier needed before next-step P1: P1 touches only buffers whose
    // last cross-phase use is >=2 barriers old.
  }
}

// ---------------------------------------------------------------------------

extern "C" void kernel_launch(void* const* d_in, const int* in_sizes, int n_in,
                              void* d_out, int out_size, void* d_ws,
                              size_t ws_size, hipStream_t stream) {
  (void)in_sizes; (void)n_in; (void)out_size; (void)ws_size;

  Params P;
  P.memory         = (const float*)d_in[0];
  P.decoder_inputs = (const float*)d_in[1];
  P.mem_len        = (const int*)  d_in[2];
  P.z_l            = (const float*)d_in[3];
  P.z_o            = (const float*)d_in[4];
  P.pre_w1 = (const float*)d_in[5];  P.pre_w2 = (const float*)d_in[6];
  P.a_wih  = (const float*)d_in[7];  P.a_whh  = (const float*)d_in[8];
  P.a_bih  = (const float*)d_in[9];  P.a_bhh  = (const float*)d_in[10];
  P.wq     = (const float*)d_in[11]; P.wm     = (const float*)d_in[12];
  P.vvec   = (const float*)d_in[13];
  P.loc_conv  = (const float*)d_in[14];
  P.loc_dense = (const float*)d_in[15];
  P.d_wih  = (const float*)d_in[16]; P.d_whh  = (const float*)d_in[17];
  P.d_bih  = (const float*)d_in[18]; P.d_bhh  = (const float*)d_in[19];
  P.res_w  = (const float*)d_in[20]; P.res_b  = (const float*)d_in[21];
  P.proj_w = (const float*)d_in[22]; P.proj_b = (const float*)d_in[23];
  P.gate_w = (const float*)d_in[24]; P.gate_b = (const float*)d_in[25];

  char* ws = (char*)d_ws;
  size_t off = 0;
  auto alloc = [&](size_t bytes) -> void* {
    void* p = ws + off;
    off += (bytes + 255) & ~(size_t)255;
    return p;
  };

  // --- zeroed region: barrier control + all recurrent state ---
  P.ctrl      = (unsigned*)alloc(256);
  P.att_h     = (float*)alloc(32 * 1024 * 4);
  P.att_c     = (float*)alloc(32 * 1024 * 4);
  P.dec_h     = (float*)alloc(32 * 1024 * 4);
  P.dec_c     = (float*)alloc(32 * 1024 * 4);
  P.aw        = (float*)alloc(32 * 256 * 4);
  P.aw_cum    = (float*)alloc(32 * 256 * 4);
  P.ctx       = (float*)alloc(32 * 512 * 4);
  P.dec_in_f  = (float*)alloc(32 * 1600 * 4);
  P.xcat      = (unsigned short*)alloc(32 * 768 * 2);
  P.att_h_bf  = (unsigned short*)alloc(32 * 1024 * 2);
  P.dec_h_bf  = (unsigned short*)alloc(32 * 1024 * 2);
  P.dec_in_bf = (unsigned short*)alloc(32 * 1600 * 2);
  size_t zero_bytes = off;

  // --- per-step scratch (fully rewritten before each read) ---
  P.gates = (float*)alloc(32 * 4096 * 4);
  P.energ = (float*)alloc(32 * 256 * 4);
  P.pq    = (float*)alloc(32 * 128 * 4);
  P.pa    = (float*)alloc((size_t)32 * 256 * 128 * 4);
  P.pm    = (float*)alloc((size_t)32 * 256 * 128 * 4);
  P.loc_bf   = (unsigned short*)alloc((size_t)32 * 32 * 256 * 2);
  P.resid_bf = (unsigned short*)alloc(32 * 1600 * 2);
  P.pre_bf   = (unsigned short*)alloc((size_t)500 * 32 * 256 * 2);

  // --- bf16 weights (converted once per call, resident in 192MB L2) ---
  P.wq_bf     = (unsigned short*)alloc((size_t)128 * 1024 * 2);
  P.ldense_bf = (unsigned short*)alloc(128 * 32 * 2);
  P.a_wih_bf  = (unsigned short*)alloc((size_t)4096 * 768 * 2);
  P.a_whh_bf  = (unsigned short*)alloc((size_t)4096 * 1024 * 2);
  P.d_wih_bf  = (unsigned short*)alloc((size_t)4096 * 1600 * 2);
  P.d_whh_bf  = (unsigned short*)alloc((size_t)4096 * 1024 * 2);
  P.res_w_bf  = (unsigned short*)alloc((size_t)1600 * 1024 * 2);
  P.proj_w_bf = (unsigned short*)alloc((size_t)80 * 1600 * 2);

  P.out_mel   = (float*)d_out;                       // [32,80,500]
  P.out_gate  = (float*)d_out + 32 * 80 * 500;       // [32,500]
  P.out_align = (float*)d_out + 32 * 80 * 500 + 32 * 500;  // [32,500,256]

  hipMemsetAsync(d_ws, 0, zero_bytes, stream);

  prenet_kernel<<<500 * 32, 256, 0, stream>>>(P);

  convert_kernel<<<512, 256, 0, stream>>>(P.wq,        P.wq_bf,     128 * 1024);
  convert_kernel<<<16,  256, 0, stream>>>(P.loc_dense, P.ldense_bf, 128 * 32);
  convert_kernel<<<2048,256, 0, stream>>>(P.a_wih,     P.a_wih_bf,  4096 * 768);
  convert_kernel<<<2048,256, 0, stream>>>(P.a_whh,     P.a_whh_bf,  4096 * 1024);
  convert_kernel<<<2048,256, 0, stream>>>(P.d_wih,     P.d_wih_bf,  4096 * 1600);
  convert_kernel<<<2048,256, 0, stream>>>(P.d_whh,     P.d_whh_bf,  4096 * 1024);
  convert_kernel<<<2048,256, 0, stream>>>(P.res_w,     P.res_w_bf,  1600 * 1024);
  convert_kernel<<<512, 256, 0, stream>>>(P.proj_w,    P.proj_w_bf, 80 * 1600);

  pm_kernel<<<2048, 256, 0, stream>>>(P);
  misc_kernel<<<32, 256, 0, stream>>>(P);

  decoder_kernel<<<NWG, NTHR, 0, stream>>>(P);
}
